// Head_88553635709649
// MI455X (gfx1250) — matrix operand order repaired and checked by reading, not compiled, not run on hardware
//
#include <hip/hip_runtime.h>
#include <hip/hip_bf16.h>

typedef _Float16 f16;
typedef __attribute__((ext_vector_type(16))) _Float16 v16h;
typedef __attribute__((ext_vector_type(8)))  _Float16 v8h;
typedef __attribute__((ext_vector_type(2)))  _Float16 h2;
typedef __attribute__((ext_vector_type(8)))  float    v8f;
typedef __attribute__((ext_vector_type(4)))  float    v4f;

union F16x16 { v16h v; _Float16 e[16]; v8h h8[2]; unsigned w[8]; };
union F32x8  { v8f  v; float    e[8];  v4f f4[2]; };
union H2U    { h2 h; unsigned u; };

// Q pre-scale: C^-0.5 * log2(e)  (softmax computed in base-2 units; exactly
// equivalent since exp2(s*log2e) == exp(s))
#define QSCALE (0.125f * 1.44269504088896340736f)

// ---------------------------------------------------------------------------
// Phase 1: q = (x@Wq)*QSCALE, k = x@Wk  -> f16 [B*T,64] row-major
//          v = x@Wv                      -> f16 [B,64,T]  (transposed)
// ---------------------------------------------------------------------------
__global__ __launch_bounds__(256) void proj_kernel(
    const float* __restrict__ x,  const float* __restrict__ Wk,
    const float* __restrict__ Wq, const float* __restrict__ Wv,
    f16* __restrict__ qf, f16* __restrict__ kf, f16* __restrict__ vt)
{
    __shared__ float xs[256];                 // 4 rows x 64 channels
    const int tid     = threadIdx.x;
    const int rowbase = blockIdx.x * 4;
    xs[tid] = x[rowbase * 64 + tid];
    __syncthreads();

    const int rl = tid >> 6;                  // local row 0..3
    const int h  = tid & 63;                  // output column
    const float* xr = &xs[rl * 64];
    float aq = 0.f, ak = 0.f, av = 0.f;
#pragma unroll
    for (int c = 0; c < 64; ++c) {
        const float xv = xr[c];
        ak += xv * Wk[c * 64 + h];
        aq += xv * Wq[c * 64 + h];
        av += xv * Wv[c * 64 + h];
    }
    const int row = rowbase + rl;             // flat B*T row
    qf[row * 64 + h] = (f16)(aq * QSCALE);
    kf[row * 64 + h] = (f16)ak;
    const int b = row >> 12;                  // row / 4096
    const int t = row & 4095;
    vt[(b * 64 + h) * 4096 + t] = (f16)av;    // transposed
}

// ---------------------------------------------------------------------------
// Fragment loaders (wave32 WMMA 16x16x32 f16 layouts)
// A (16x32, MxK): lane m=l&15, half=l>>4; VGPRs cover K = 16*half + {0..15}
//                 -> two 16B loads at +0 and +8 elems
// B (32x16, KxN): lane n=l&15, half=l>>4; VGPRs cover K = 16*half + {0..15}
//                 -> 32 contiguous bytes at n*ld + 16*half
// ---------------------------------------------------------------------------
__device__ __forceinline__ v16h load_frag_a(const f16* __restrict__ src, int ld) {
    const int lane = threadIdx.x & 31;
    const int m  = lane & 15;
    const int lh = lane >> 4;
    const f16* p = src + m * ld + lh * 16;
    F16x16 u;
    u.h8[0] = *(const v8h*)(p);
    u.h8[1] = *(const v8h*)(p + 8);
    return u.v;
}
__device__ __forceinline__ v16h load_frag_b(const f16* __restrict__ src, int ld) {
    const int lane = threadIdx.x & 31;
    const int n  = lane & 15;
    const int lh = lane >> 4;
    const f16* p = src + n * ld + lh * 16;
    F16x16 u;
    u.h8[0] = *(const v8h*)(p);
    u.h8[1] = *(const v8h*)(p + 8);
    return u.v;
}

// ---------------------------------------------------------------------------
// Phase 2: flash attention. One wave owns 16 query rows.
// S^T(64s x 16q) = K_tile @ Q^T ; online softmax over s; O^T += V^T @ P^T.
// ---------------------------------------------------------------------------
__global__ __launch_bounds__(256) void attn_kernel(
    const f16* __restrict__ qf, const f16* __restrict__ kf,
    const f16* __restrict__ vt, float* __restrict__ out)
{
    const int lane  = threadIdx.x & 31;
    const int wave  = threadIdx.x >> 5;
    const int strip = blockIdx.x * 8 + wave;   // 0..2047
    const int b     = strip >> 8;              // batch
    const int q0    = (strip & 255) << 4;      // first query row of strip
    const int qlane = lane & 15;               // this lane's query column
    const int lh    = lane >> 4;               // lane half

    const f16* qb = qf + ((size_t)(b * 4096 + q0)) * 64;
    const f16* kb = kf + (size_t)b * 4096 * 64;
    const f16* vb = vt + (size_t)b * 64 * 4096;

    // Q^T as B fragments (contraction over h = 64 -> 2 fragments), persistent
    const v16h qt0 = load_frag_b(qb, 64);
    const v16h qt1 = load_frag_b(qb + 32, 64);

    F32x8 o[4];
#pragma unroll
    for (int i = 0; i < 4; ++i) o[i].v = (v8f){};
    float mrun = -1e30f, lsum = 0.f;

    const int lastkt = q0 >> 6;
    for (int kt = 0; kt <= lastkt; ++kt) {
        const int kbase = kt << 6;

        // ---- S^T tiles: 4 stacked 16x16 tiles over s ----
        F32x8 s[4];
#pragma unroll
        for (int st = 0; st < 4; ++st) {
            const f16* krow = kb + (size_t)(kbase + st * 16) * 64;
            const v16h ka0 = load_frag_a(krow, 64);
            const v16h ka1 = load_frag_a(krow + 32, 64);
            v8f acc = (v8f){};
            acc = __builtin_amdgcn_wmma_f32_16x16x32_f16(false, ka0, false, qt0,
                                                         (short)0, acc, false, false);
            acc = __builtin_amdgcn_wmma_f32_16x16x32_f16(false, ka1, false, qt1,
                                                         (short)0, acc, false, false);
            s[st].v = acc;
        }

        // ---- causal mask (diagonal tile only) ----
        if (kt == lastkt) {
            const int qg = q0 + qlane;
#pragma unroll
            for (int st = 0; st < 4; ++st)
#pragma unroll
                for (int r = 0; r < 8; ++r) {
                    const int sg = kbase + st * 16 + r + lh * 8;
                    if (sg > qg) s[st].e[r] = -1e30f;
                }
        }

        // ---- online softmax stats (per query = per lane, halves merged) ----
        float sm = -1e30f;
#pragma unroll
        for (int st = 0; st < 4; ++st)
#pragma unroll
            for (int r = 0; r < 8; ++r) sm = fmaxf(sm, s[st].e[r]);
        sm = fmaxf(sm, __shfl_xor(sm, 16, 32));
        const float mn    = fmaxf(mrun, sm);
        const float alpha = exp2f(mrun - mn);    // base-2 units
        mrun = mn;

        float rs = 0.f;
#pragma unroll
        for (int st = 0; st < 4; ++st)
#pragma unroll
            for (int r = 0; r < 8; ++r) {
                const float p = exp2f(s[st].e[r] - mn);
                s[st].e[r] = p;
                rs += p;
            }
        rs += __shfl_xor(rs, 16, 32);
        lsum = lsum * alpha + rs;

        // ---- rescale accumulator ----
#pragma unroll
        for (int i = 0; i < 4; ++i)
#pragma unroll
            for (int r = 0; r < 8; ++r) o[i].e[r] *= alpha;

        // ---- P^T repack: pack to f16 pairs first, then exchange only what the
        //      partner half actually consumes (8 shuffles per K-tile) ----
        unsigned pk[4][4];
#pragma unroll
        for (int t = 0; t < 4; ++t)
#pragma unroll
            for (int p = 0; p < 4; ++p) {
                H2U u2;
                u2.h.x = (f16)s[t].e[2 * p];
                u2.h.y = (f16)s[t].e[2 * p + 1];
                pk[t][p] = u2.u;
            }
        // lh==0 lanes need partner's tiles {0,2}; lh==1 lanes need tiles {1,3}.
        // Each lane sends what its partner needs.
        unsigned recvA[4], recvB[4];
#pragma unroll
        for (int p = 0; p < 4; ++p) {
            const unsigned sA = lh ? pk[0][p] : pk[1][p];
            const unsigned sB = lh ? pk[2][p] : pk[3][p];
            recvA[p] = (unsigned)__shfl_xor((int)sA, 16, 32);
            recvB[p] = (unsigned)__shfl_xor((int)sB, 16, 32);
        }
        F16x16 pb[2];
#pragma unroll
        for (int p = 0; p < 4; ++p) {
            pb[0].w[p]     = lh ? recvA[p] : pk[0][p];
            pb[0].w[p + 4] = lh ? pk[1][p] : recvA[p];
            pb[1].w[p]     = lh ? recvB[p] : pk[2][p];
            pb[1].w[p + 4] = lh ? pk[3][p] : recvB[p];
        }

        // ---- O^T += V^T @ P^T ----
#pragma unroll
        for (int ht = 0; ht < 4; ++ht) {
            const f16* vrow = vb + (size_t)(ht * 16) * 4096 + kbase;
            const v16h va0 = load_frag_a(vrow, 4096);
            const v16h va1 = load_frag_a(vrow + 32, 4096);
            v8f acc = o[ht].v;
            acc = __builtin_amdgcn_wmma_f32_16x16x32_f16(false, va0, false, pb[0].v,
                                                         (short)0, acc, false, false);
            acc = __builtin_amdgcn_wmma_f32_16x16x32_f16(false, va1, false, pb[1].v,
                                                         (short)0, acc, false, false);
            o[ht].v = acc;
        }
    }

    // ---- normalize + store: lane holds (h = r + 8*lh + 16*ht, q = qlane) ----
    const float inv = 1.0f / lsum;
    float* orow = out + ((size_t)(b * 4096 + q0 + qlane)) * 64;
#pragma unroll
    for (int ht = 0; ht < 4; ++ht) {
#pragma unroll
        for (int r = 0; r < 8; ++r) o[ht].e[r] *= inv;
        const int h0 = ht * 16 + lh * 8;
        *(v4f*)(orow + h0)     = o[ht].f4[0];
        *(v4f*)(orow + h0 + 4) = o[ht].f4[1];
    }
}

// ---------------------------------------------------------------------------
extern "C" void kernel_launch(void* const* d_in, const int* in_sizes, int n_in,
                              void* d_out, int out_size, void* d_ws, size_t ws_size,
                              hipStream_t stream) {
    (void)in_sizes; (void)n_in; (void)out_size; (void)ws_size;
    const float* x  = (const float*)d_in[0];
    const float* Wk = (const float*)d_in[1];
    const float* Wq = (const float*)d_in[2];
    const float* Wv = (const float*)d_in[3];
    float* out = (float*)d_out;

    const size_t NTOK = (size_t)8 * 4096 * 64;   // B*T*H elements
    f16* qf = (f16*)d_ws;
    f16* kf = qf + NTOK;
    f16* vt = kf + NTOK;                         // [B,64,T]

    // 8192 blocks x 256 threads: 4 rows per block
    proj_kernel<<<8192, 256, 0, stream>>>(x, Wk, Wq, Wv, qf, kf, vt);
    // 2048 strips of 16 queries, 8 waves per block
    attn_kernel<<<256, 256, 0, stream>>>(qf, kf, vt, out);
}
